// Pointnet2Backbone_p_15290083574260
// MI455X (gfx1250) — matrix operand order repaired and checked by reading, not compiled, not run on hardware
//
#include <hip/hip_runtime.h>

#define BB 4
#define NN 16384

typedef __attribute__((ext_vector_type(16))) __bf16           v16bf;
typedef __attribute__((ext_vector_type(16))) unsigned short   v16us;
typedef __attribute__((ext_vector_type(8)))  float            v8f;

__device__ __forceinline__ unsigned short f2bf(float f) {
  unsigned u = __float_as_uint(f);
  u += 0x7fffu + ((u >> 16) & 1u);
  return (unsigned short)(u >> 16);
}
__device__ __forceinline__ float bf2f(unsigned short h) {
  return __uint_as_float(((unsigned)h) << 16);
}

// ---------------------------------------------------------------------------
// Small utility kernels
// ---------------------------------------------------------------------------
__global__ void k_extract_xyz(const float* __restrict__ pc, float* __restrict__ xyz, int total /*B*N*/) {
  int g = blockIdx.x * blockDim.x + threadIdx.x;
  if (g >= total) return;
  xyz[(size_t)g * 3 + 0] = pc[(size_t)g * 4 + 0];
  xyz[(size_t)g * 3 + 1] = pc[(size_t)g * 4 + 1];
  xyz[(size_t)g * 3 + 2] = pc[(size_t)g * 4 + 2];
}

__global__ void k_paint(const float* __restrict__ pc, int* __restrict__ painted, int total) {
  int g = blockIdx.x * blockDim.x + threadIdx.x;
  if (g >= total) return;
  painted[g] = (int)pc[(size_t)g * 4 + 3];
}

__global__ void k_umb(const float* __restrict__ rs, const float* __restrict__ W,
                      const float* __restrict__ bias, float* __restrict__ out, int total) {
  int g = blockIdx.x * blockDim.x + threadIdx.x;
  if (g >= total) return;
  const float* x = rs + (size_t)g * 10;
  float* y = out + (size_t)g * 10;
  for (int j = 0; j < 10; ++j) {
    float a = bias[j];
    for (int k = 0; k < 10; ++k) a += x[k] * W[k * 10 + j];
    y[j] = fmaxf(a, 0.f);
  }
}

__global__ void k_f2bf(const float* __restrict__ s, unsigned short* __restrict__ d, int n) {
  int g = blockIdx.x * blockDim.x + threadIdx.x;
  if (g < n) d[g] = f2bf(s[g]);
}

__global__ void k_i2f(const int* __restrict__ s, float* __restrict__ d, int n) {
  int g = blockIdx.x * blockDim.x + threadIdx.x;
  if (g < n) d[g] = (float)s[g];
}

__global__ void k_zero_i(int* __restrict__ p, int n) {
  int g = blockIdx.x * blockDim.x + threadIdx.x;
  if (g < n) p[g] = 0;
}

__global__ void k_scatter1(int* __restrict__ mask, const int* __restrict__ inds,
                           int npoint, int n, int total /*B*npoint*/) {
  int g = blockIdx.x * blockDim.x + threadIdx.x;
  if (g >= total) return;
  int b = g / npoint;
  mask[(size_t)b * n + inds[g]] = 1;
}

// dst[b, off+i, c] = src[b, i, c] with dst row-pitch `pitch`
__global__ void k_copy_rows_f(const float* __restrict__ src, float* __restrict__ dst,
                              int rows, int C, int pitch, int off, int total) {
  int g = blockIdx.x * blockDim.x + threadIdx.x;
  if (g >= total) return;
  int c = g % C, t = g / C, i = t % rows, b = t / rows;
  dst[((size_t)b * pitch + off + i) * C + c] = src[g];
}

__global__ void k_copy_i(const int* __restrict__ src, int* __restrict__ dst,
                         int rows, int pitch, int off, int total) {
  int g = blockIdx.x * blockDim.x + threadIdx.x;
  if (g >= total) return;
  int i = g % rows, b = g / rows;
  dst[(size_t)b * pitch + off + i] = src[g];
}

// dst[b, i, c] = src[b, idx[b,i], c]
__global__ void k_gather_rows_f(const float* __restrict__ src, int n_src,
                                const int* __restrict__ idx, int n_idx, int C,
                                float* __restrict__ dst, int total) {
  int g = blockIdx.x * blockDim.x + threadIdx.x;
  if (g >= total) return;
  int c = g % C, t = g / C, i = t % n_idx, b = t / n_idx;
  int j = idx[(size_t)b * n_idx + i];
  dst[g] = src[((size_t)b * n_src + j) * C + c];
}

__global__ void k_gather_i(const int* __restrict__ src, int n_src,
                           const int* __restrict__ idx, int n_idx,
                           int* __restrict__ dst, int total) {
  int g = blockIdx.x * blockDim.x + threadIdx.x;
  if (g >= total) return;
  int i = g % n_idx, b = g / n_idx;
  dst[g] = src[(size_t)b * n_src + idx[(size_t)b * n_idx + i]];
}

// ---------------------------------------------------------------------------
// Farthest point sampling: one 1024-thread workgroup per batch element.
// ---------------------------------------------------------------------------
__global__ __launch_bounds__(1024) void k_fps(const float* __restrict__ xyz,
                                              const int* __restrict__ painted,
                                              float wght, int n, int npoint,
                                              int* __restrict__ inds) {
  int b = blockIdx.x;
  xyz += (size_t)b * n * 3;
  painted += (size_t)b * n;
  inds += (size_t)b * npoint;
  const int T = 1024;
  int tid = threadIdx.x, lane = tid & 31, wid = tid >> 5;

  float px[16], py[16], pz[16], wv[16], dist[16];
  int cnt = 0;
  for (int i = tid; i < n; i += T) {
    px[cnt] = xyz[(size_t)i * 3 + 0];
    py[cnt] = xyz[(size_t)i * 3 + 1];
    pz[cnt] = xyz[(size_t)i * 3 + 2];
    wv[cnt] = 1.0f + (wght - 1.0f) * (float)painted[i];
    dist[cnt] = 1e10f;
    ++cnt;
  }

  __shared__ float s_val[32];
  __shared__ int   s_idx[32];
  __shared__ float s_pt[3];
  __shared__ int   s_last;
  if (tid == 0) {
    s_last = 0;
    s_pt[0] = xyz[0]; s_pt[1] = xyz[1]; s_pt[2] = xyz[2];
  }
  __syncthreads();

  for (int it = 0; it < npoint; ++it) {
    float lx = s_pt[0], ly = s_pt[1], lz = s_pt[2];
    if (tid == 0) inds[it] = s_last;

    float bestv = -1.0f; int besti = -1;
    for (int c = 0; c < cnt; ++c) {
      float dx = px[c] - lx, dy = py[c] - ly, dz = pz[c] - lz;
      float d = (dx * dx + dy * dy + dz * dz) * wv[c];
      float nd = fminf(dist[c], d);
      dist[c] = nd;
      int gi = tid + c * T;
      if (nd > bestv || (nd == bestv && gi < besti)) { bestv = nd; besti = gi; }
    }
    for (int off = 16; off > 0; off >>= 1) {
      float ov = __shfl_xor(bestv, off);
      int   oi = __shfl_xor(besti, off);
      if (oi >= 0 && (ov > bestv || (ov == bestv && (besti < 0 || oi < besti)))) { bestv = ov; besti = oi; }
    }
    if (lane == 0) { s_val[wid] = bestv; s_idx[wid] = besti; }
    __syncthreads();
    if (tid < 32) {
      float v = s_val[tid]; int ii = s_idx[tid];
      for (int off = 16; off > 0; off >>= 1) {
        float ov = __shfl_xor(v, off);
        int   oi = __shfl_xor(ii, off);
        if (oi >= 0 && (ov > v || (ov == v && (ii < 0 || oi < ii)))) { v = ov; ii = oi; }
      }
      if (tid == 0) {
        if (ii < 0) ii = 0;
        s_last = ii;
        s_pt[0] = xyz[(size_t)ii * 3 + 0];
        s_pt[1] = xyz[(size_t)ii * 3 + 1];
        s_pt[2] = xyz[(size_t)ii * 3 + 2];
      }
    }
    __syncthreads();
  }
}

// ---------------------------------------------------------------------------
// Stable compaction of unsampled points (remove_sampled)
// ---------------------------------------------------------------------------
__global__ __launch_bounds__(1024) void k_compact(const int* __restrict__ mask, int n,
                                                  int keep, int* __restrict__ order) {
  int b = blockIdx.x;
  mask += (size_t)b * n;
  order += (size_t)b * keep;
  __shared__ int s[1024];
  __shared__ int base;
  if (threadIdx.x == 0) base = 0;
  __syncthreads();
  for (int c0 = 0; c0 < n; c0 += 1024) {
    int i = c0 + threadIdx.x;
    int f = (i < n) ? (mask[i] == 0 ? 1 : 0) : 0;
    s[threadIdx.x] = f;
    __syncthreads();
    for (int off = 1; off < 1024; off <<= 1) {
      int v = (threadIdx.x >= off) ? s[threadIdx.x - off] : 0;
      __syncthreads();
      s[threadIdx.x] += v;
      __syncthreads();
    }
    if (f) {
      int pos = base + s[threadIdx.x] - 1;
      if (pos < keep) order[pos] = i;
    }
    __syncthreads();
    if (threadIdx.x == 1023) base += s[1023];
    __syncthreads();
  }
}

// ---------------------------------------------------------------------------
// Ball query: k-nearest within radius, slots filled with nearest
// ---------------------------------------------------------------------------
template <int NS>
__global__ void k_ball(const float* __restrict__ centers, int npoint,
                       const float* __restrict__ pts, int n, float r2,
                       int* __restrict__ idx, int total) {
  int g = blockIdx.x * blockDim.x + threadIdx.x;
  if (g >= total) return;
  int b = g / npoint;
  const float* c = centers + (size_t)g * 3;
  const float* P = pts + (size_t)b * n * 3;
  float bd[NS]; int bi[NS];
#pragma unroll
  for (int s = 0; s < NS; ++s) { bd[s] = 1e10f; bi[s] = -1; }
  float cx = c[0], cy = c[1], cz = c[2];
  for (int j = 0; j < n; ++j) {
    float dx = P[(size_t)j * 3 + 0] - cx;
    float dy = P[(size_t)j * 3 + 1] - cy;
    float dz = P[(size_t)j * 3 + 2] - cz;
    float d = dx * dx + dy * dy + dz * dz;
    if (d <= r2 && d < bd[NS - 1]) {
      int pos = NS - 1;
      while (pos > 0 && bd[pos - 1] > d) {
        bd[pos] = bd[pos - 1]; bi[pos] = bi[pos - 1]; --pos;
      }
      bd[pos] = d; bi[pos] = j;
    }
  }
  int fill = (bi[0] < 0) ? 0 : bi[0];
  int* o = idx + (size_t)g * NS;
#pragma unroll
  for (int s = 0; s < NS; ++s) o[s] = (bi[s] < 0) ? fill : bi[s];
}

// ---------------------------------------------------------------------------
// Grouped feature assembly -> bf16 rows of stride Kpad (zero padded)
// ---------------------------------------------------------------------------
__global__ void k_group(const float* __restrict__ centers, const int* __restrict__ idx,
                        const float* __restrict__ pts, const float* __restrict__ feats,
                        int n, int C, int npoint, int nsample, float radius, int Kpad,
                        unsigned short* __restrict__ X, int total) {
  int g = blockIdx.x * blockDim.x + threadIdx.x;
  if (g >= total) return;
  int p = (g / nsample) % npoint;
  int b = g / (nsample * npoint);
  int j = idx[g];
  const float* pc = centers + ((size_t)b * npoint + p) * 3;
  const float* pj = pts + ((size_t)b * n + j) * 3;
  float rx = pj[0] - pc[0], ry = pj[1] - pc[1], rz = pj[2] - pc[2];
  float rr = sqrtf(rx * rx + ry * ry + rz * rz + 1e-12f);
  float ct = rz / rr;
  ct = fminf(fmaxf(ct, -1.f), 1.f);
  float theta = acosf(ct);
  float phi = atan2f(ry, rx);
  unsigned short* row = X + (size_t)g * Kpad;
  row[0] = f2bf(rr); row[1] = f2bf(theta); row[2] = f2bf(phi);
  float inv = 1.0f / radius;
  row[3] = f2bf(rx * inv); row[4] = f2bf(ry * inv); row[5] = f2bf(rz * inv);
  const float* f = feats + ((size_t)b * n + j) * C;
  for (int c = 0; c < C; ++c) row[6 + c] = f2bf(f[c]);
  for (int c = 6 + C; c < Kpad; ++c) row[c] = 0;  // zero pad -> GEMM lda path
}

// ---------------------------------------------------------------------------
// bf16 WMMA GEMM: Y = act(X[MxK, row stride lda] * W[KxN] + bias).
// 64x64 block tile, 4 waves * (16 rows x 64 cols), K stepped by 32.
// lda % 8 == 0 always (padded inputs), so staging is pure uint4 loads.
// LDS tiles stored in WMMA fragment order: each lane's 16 bf16 fragment is 32
// contiguous bytes -> two ds_load_b128 per fragment on the critical path.
// ---------------------------------------------------------------------------
__global__ __launch_bounds__(128) void k_gemm_bias_act(
    const unsigned short* __restrict__ X, const unsigned short* __restrict__ W,
    const float* __restrict__ bias, unsigned short* __restrict__ Yb,
    float* __restrict__ Yf, int M, int K, int lda, int N, int relu) {
  // A fragments: [wave 0..3][lane 0..31][e 0..15]
  __shared__ __align__(32) unsigned short Xf[4 * 32 * 16];
  // B fragments: [colgroup 0..3][lane 0..31][e 0..15]
  __shared__ __align__(32) unsigned short Wf[4 * 32 * 16];

  int tile_m = blockIdx.x * 64;
  int tile_n = blockIdx.y * 64;
  int tid = threadIdx.x;
  int lane = tid & 31, wave = tid >> 5;
  int cl = lane & 15, hi = lane >> 4;

  v8f acc[4] = {};

  for (int k0 = 0; k0 < K; k0 += 32) {
    __syncthreads();

    // ---- stage X tile (64 rows x 32 k) into A-fragment layout ----
    for (int t = tid; t < 256; t += 128) {
      int r = t >> 2, k8 = (t & 3) << 3;
      int gm = tile_m + r, gk = k0 + k8;
      uint4 v = make_uint4(0u, 0u, 0u, 0u);
      if (gm < M && gk < lda) v = *(const uint4*)(X + (size_t)gm * lda + gk);
      unsigned short es[8];
      es[0] = (unsigned short)(v.x & 0xffffu); es[1] = (unsigned short)(v.x >> 16);
      es[2] = (unsigned short)(v.y & 0xffffu); es[3] = (unsigned short)(v.y >> 16);
      es[4] = (unsigned short)(v.z & 0xffffu); es[5] = (unsigned short)(v.z >> 16);
      es[6] = (unsigned short)(v.w & 0xffffu); es[7] = (unsigned short)(v.w >> 16);
      int w = r >> 4, m = r & 15;
#pragma unroll
      for (int q = 0; q < 8; ++q) {
        int k = k8 + q;
        int khi = (k >> 3) & 1;
        int ee = (k & 7) + ((k >> 4) << 3);
        Xf[(((w << 5) + (khi << 4) + m) << 4) + ee] = es[q];
      }
    }

    // ---- stage W tile (32 k x 64 n) into B-fragment layout ----
    for (int t = tid; t < 256; t += 128) {
      int r = t >> 3, c8 = (t & 7) << 3;
      int gk = k0 + r;
      uint4 v = make_uint4(0u, 0u, 0u, 0u);
      if (gk < K) v = *(const uint4*)(W + (size_t)gk * N + tile_n + c8);
      unsigned short es[8];
      es[0] = (unsigned short)(v.x & 0xffffu); es[1] = (unsigned short)(v.x >> 16);
      es[2] = (unsigned short)(v.y & 0xffffu); es[3] = (unsigned short)(v.y >> 16);
      es[4] = (unsigned short)(v.z & 0xffffu); es[5] = (unsigned short)(v.z >> 16);
      es[6] = (unsigned short)(v.w & 0xffffu); es[7] = (unsigned short)(v.w >> 16);
      int khi = (r >> 3) & 1, e = (r & 7) + ((r >> 4) << 3);
#pragma unroll
      for (int q = 0; q < 8; ++q) {
        int c = c8 + q;
        int g4 = c >> 4, ccl = c & 15;
        Wf[(((g4 << 5) + (khi << 4) + ccl) << 4) + e] = es[q];
      }
    }

    if (k0 + 32 < K) {
      // stream-prefetch next A panel (global_prefetch_b8)
      __builtin_prefetch(X + (size_t)(tile_m + (tid >> 1)) * lda + k0 + 32, 0, 1);
    }
    __syncthreads();

    // ---- fragments: two ds_load_b128 each ----
    v16us au = *(const v16us*)&Xf[((wave << 5) + lane) << 4];
    v16bf a = __builtin_bit_cast(v16bf, au);
#pragma unroll
    for (int g4 = 0; g4 < 4; ++g4) {
      v16us bu = *(const v16us*)&Wf[((g4 << 5) + lane) << 4];
      v16bf bb = __builtin_bit_cast(v16bf, bu);
      acc[g4] = __builtin_amdgcn_wmma_f32_16x16x32_bf16(
          false, a, false, bb, (short)0, acc[g4], false, false);
    }
  }

#pragma unroll
  for (int g4 = 0; g4 < 4; ++g4) {
    int gn = tile_n + g4 * 16 + cl;
    float bsv = bias ? bias[gn] : 0.f;
#pragma unroll
    for (int v = 0; v < 8; ++v) {
      int gm = tile_m + wave * 16 + v + hi * 8;  // C/D layout: lanes16-31 -> M=v+8
      if (gm < M) {
        float y = acc[g4][v] + bsv;
        if (relu) y = fmaxf(y, 0.f);
        Yb[(size_t)gm * N + gn] = f2bf(y);
        if (Yf) Yf[(size_t)gm * N + gn] = y;
      }
    }
  }
}

// ---------------------------------------------------------------------------
// Max pool over nsample (bf16 in, f32 out)
// ---------------------------------------------------------------------------
__global__ void k_maxpool(const unsigned short* __restrict__ Y, int nsample, int C,
                          float* __restrict__ out, int total /*B*npoint*C*/) {
  int g = blockIdx.x * blockDim.x + threadIdx.x;
  if (g >= total) return;
  int c = g % C, bp = g / C;
  const unsigned short* p = Y + (size_t)bp * nsample * C + c;
  float m = -1e30f;
  for (int s = 0; s < nsample; ++s) m = fmaxf(m, bf2f(p[(size_t)s * C]));
  out[g] = m;
}

// ---------------------------------------------------------------------------
// FP: 3-NN inverse-distance interpolation + concat -> bf16 rows of 2C
// ---------------------------------------------------------------------------
__global__ void k_fp_build(const float* __restrict__ fxyz, int nf,
                           const float* __restrict__ cxyz, int nc,
                           const float* __restrict__ ff, const float* __restrict__ cf,
                           int C, unsigned short* __restrict__ X, int total /*B*nf*/) {
  int g = blockIdx.x * blockDim.x + threadIdx.x;
  if (g >= total) return;
  int b = g / nf;
  const float* p = fxyz + (size_t)g * 3;
  const float* Cx = cxyz + (size_t)b * nc * 3;
  float px = p[0], py = p[1], pz = p[2];
  float d0 = 1e30f, d1 = 1e30f, d2 = 1e30f;
  int i0 = 0, i1 = 0, i2 = 0;
  for (int j = 0; j < nc; ++j) {
    float dx = Cx[(size_t)j * 3 + 0] - px;
    float dy = Cx[(size_t)j * 3 + 1] - py;
    float dz = Cx[(size_t)j * 3 + 2] - pz;
    float d = dx * dx + dy * dy + dz * dz;
    if (d < d0)      { d2 = d1; i2 = i1; d1 = d0; i1 = i0; d0 = d; i0 = j; }
    else if (d < d1) { d2 = d1; i2 = i1; d1 = d; i1 = j; }
    else if (d < d2) { d2 = d; i2 = j; }
  }
  float w0 = 1.f / (d0 + 1e-8f), w1 = 1.f / (d1 + 1e-8f), w2 = 1.f / (d2 + 1e-8f);
  float ws = w0 + w1 + w2;
  w0 /= ws; w1 /= ws; w2 /= ws;
  unsigned short* row = X + (size_t)g * (2 * C);
  const float* f = ff + (size_t)g * C;
  for (int c = 0; c < C; ++c) row[c] = f2bf(f[c]);
  const float* c0 = cf + ((size_t)b * nc + i0) * C;
  const float* c1 = cf + ((size_t)b * nc + i1) * C;
  const float* c2 = cf + ((size_t)b * nc + i2) * C;
  for (int c = 0; c < C; ++c)
    row[C + c] = f2bf(w0 * c0[c] + w1 * c1[c] + w2 * c2[c]);
}

// ---------------------------------------------------------------------------
// Host orchestration
// ---------------------------------------------------------------------------
static inline int g256(int n) { return (n + 255) / 256; }

extern "C" void kernel_launch(void* const* d_in, const int* in_sizes, int n_in,
                              void* d_out, int out_size, void* d_ws, size_t ws_size,
                              hipStream_t stream) {
  (void)in_sizes; (void)n_in; (void)out_size; (void)ws_size;

  const float* d_pc = (const float*)d_in[0];
  const float* d_rs = (const float*)d_in[1];
  int pi = 2;
  const float* umbW = (const float*)d_in[pi++];
  const float* umbB = (const float*)d_in[pi++];
  const float *saW[4][3], *saBias[4][3];
  for (int s = 0; s < 4; ++s)
    for (int l = 0; l < 3; ++l) {
      saW[s][l]    = (const float*)d_in[pi++];
      saBias[s][l] = (const float*)d_in[pi++];
    }
  const float *fpW[2][2], *fpBias[2][2];
  for (int s = 0; s < 2; ++s)
    for (int l = 0; l < 2; ++l) {
      fpW[s][l]    = (const float*)d_in[pi++];
      fpBias[s][l] = (const float*)d_in[pi++];
    }

  // output layout
  float* out_fp2f    = (float*)d_out;                  // 4*1024*256
  float* out_sa2xyz  = out_fp2f + 4 * 1024 * 256;      // 4*1024*3
  float* out_fp2inds = out_sa2xyz + 4 * 1024 * 3;      // 4*1024
  float* out_sa1xyz  = out_fp2inds + 4 * 1024;         // 4*2048*3
  float* out_sa1inds = out_sa1xyz + 4 * 2048 * 3;      // 4*2048

  // workspace bump allocator
  char* wsbase = (char*)d_ws;
  size_t off = 0;
  auto alloc = [&](size_t bytes) -> void* {
    off = (off + 255) & ~(size_t)255;
    void* p = wsbase + off;
    off += bytes;
    return p;
  };

  int*   painted = (int*)alloc((size_t)BB * NN * 4);
  float* xyz0    = (float*)alloc((size_t)BB * NN * 3 * 4);
  float* feats0  = (float*)alloc((size_t)BB * NN * 10 * 4);
  int*   mask    = (int*)alloc((size_t)BB * NN * 4);
  int*   order   = (int*)alloc((size_t)BB * 15360 * 4);
  float* nxyz    = (float*)alloc((size_t)BB * 15360 * 3 * 4);
  int*   npnt    = (int*)alloc((size_t)BB * 15360 * 4);
  float* nfeat   = (float*)alloc((size_t)BB * 15360 * 10 * 4);
  int*   idxbuf  = (int*)alloc((size_t)BB * 1024 * 64 * 4);
  unsigned short* bufG = (unsigned short*)alloc((size_t)9500000 * 2);
  unsigned short* bufA = (unsigned short*)alloc((size_t)33554432 * 2);
  unsigned short* bufB = (unsigned short*)alloc((size_t)33554432 * 2);

  int *sa1_i1 = (int*)alloc(BB * 1024 * 4), *sa1_i2 = (int*)alloc(BB * 1024 * 4);
  float *c11 = (float*)alloc(BB * 1024 * 3 * 4), *c12 = (float*)alloc(BB * 1024 * 3 * 4);
  int *sa1_p1 = (int*)alloc(BB * 1024 * 4), *sa1_p2 = (int*)alloc(BB * 1024 * 4);
  float *sa1_f1 = (float*)alloc((size_t)BB * 1024 * 128 * 4);
  float *sa1_f2 = (float*)alloc((size_t)BB * 1024 * 128 * 4);
  float *sa1_f  = (float*)alloc((size_t)BB * 2048 * 128 * 4);

  int *sa2_i1 = (int*)alloc(BB * 512 * 4), *sa2_i2 = (int*)alloc(BB * 512 * 4);
  float *c21 = (float*)alloc(BB * 512 * 3 * 4), *c22 = (float*)alloc(BB * 512 * 3 * 4);
  int *sa2_p1 = (int*)alloc(BB * 512 * 4), *sa2_p2 = (int*)alloc(BB * 512 * 4);
  float *sa2_f1 = (float*)alloc((size_t)BB * 512 * 256 * 4);
  float *sa2_f2 = (float*)alloc((size_t)BB * 512 * 256 * 4);
  float *sa2_f  = (float*)alloc((size_t)BB * 1024 * 256 * 4);

  int *sa3_i1 = (int*)alloc(BB * 256 * 4), *sa3_i2 = (int*)alloc(BB * 256 * 4);
  float *c31 = (float*)alloc(BB * 256 * 3 * 4), *c32 = (float*)alloc(BB * 256 * 3 * 4);
  int *sa3_p1 = (int*)alloc(BB * 256 * 4), *sa3_p2 = (int*)alloc(BB * 256 * 4);
  float *sa3_f1 = (float*)alloc((size_t)BB * 256 * 256 * 4);
  float *sa3_f2 = (float*)alloc((size_t)BB * 256 * 256 * 4);
  float *sa3_xyz = (float*)alloc(BB * 512 * 3 * 4);
  float *sa3_f   = (float*)alloc((size_t)BB * 512 * 256 * 4);
  int   *sa3_p   = (int*)alloc(BB * 512 * 4);

  int *sa4_i = (int*)alloc(BB * 256 * 4);
  float *c4 = (float*)alloc(BB * 256 * 3 * 4);
  int *sa4_p = (int*)alloc(BB * 256 * 4);
  float *sa4_f = (float*)alloc((size_t)BB * 256 * 256 * 4);
  float *fp1_f = (float*)alloc((size_t)BB * 512 * 256 * 4);

  int *seed1 = (int*)alloc(BB * 512 * 4), *sa12 = (int*)alloc(BB * 512 * 4);
  int *seed2 = (int*)alloc(BB * 512 * 4), *i2orig = (int*)alloc(BB * 1024 * 4);
  int *sa1_inds_i = (int*)alloc(BB * 2048 * 4), *fp2_inds_i = (int*)alloc(BB * 1024 * 4);

  // bf16 weight conversion
  auto conv = [&](const float* Wm, int k, int n) -> unsigned short* {
    unsigned short* p = (unsigned short*)alloc((size_t)k * n * 2);
    int tot = k * n;
    k_f2bf<<<g256(tot), 256, 0, stream>>>(Wm, p, tot);
    return p;
  };
  const int saDims[4][4] = {{16, 64, 64, 128},
                            {134, 128, 128, 256},
                            {262, 128, 128, 256},
                            {262, 128, 128, 256}};
  unsigned short* saWb[4][3];
  for (int s = 0; s < 4; ++s)
    for (int l = 0; l < 3; ++l)
      saWb[s][l] = conv(saW[s][l], saDims[s][l], saDims[s][l + 1]);
  unsigned short* fpWb[2][2];
  for (int s = 0; s < 2; ++s) {
    fpWb[s][0] = conv(fpW[s][0], 512, 256);
    fpWb[s][1] = conv(fpW[s][1], 256, 256);
  }

  auto gemm = [&](const unsigned short* X, const unsigned short* Wm, const float* bias,
                  unsigned short* Yb, float* Yf, int M, int K, int lda, int N) {
    dim3 grid((M + 63) / 64, N / 64);
    k_gemm_bias_act<<<grid, 128, 0, stream>>>(X, Wm, bias, Yb, Yf, M, K, lda, N, 1);
  };

  auto ball = [&](const float* centers, int np, const float* pts, int nb, float rad, int ns) {
    int tot = BB * np;
    dim3 grid((tot + 127) / 128);
    float r2 = rad * rad;
    if (ns == 64)      k_ball<64><<<grid, 128, 0, stream>>>(centers, np, pts, nb, r2, idxbuf, tot);
    else if (ns == 32) k_ball<32><<<grid, 128, 0, stream>>>(centers, np, pts, nb, r2, idxbuf, tot);
    else               k_ball<16><<<grid, 128, 0, stream>>>(centers, np, pts, nb, r2, idxbuf, tot);
  };

  auto run_sa = [&](const float* xyzf, const int* pf, int nf,
                    const float* ptsb, const float* fb, int nb, int C,
                    int np, float rad, int ns, float wght, int sidx,
                    int* inds, float* centers, int* pout, float* fout) {
    k_fps<<<BB, 1024, 0, stream>>>(xyzf, pf, wght, nf, np, inds);
    int tot = BB * np * 3;
    k_gather_rows_f<<<g256(tot), 256, 0, stream>>>(xyzf, nf, inds, np, 3, centers, tot);
    tot = BB * np;
    k_gather_i<<<g256(tot), 256, 0, stream>>>(pf, nf, inds, np, pout, tot);
    ball(centers, np, ptsb, nb, rad, ns);
    int K = 6 + C;
    int Kpad = (K + 7) & ~7;
    tot = BB * np * ns;
    k_group<<<g256(tot), 256, 0, stream>>>(centers, idxbuf, ptsb, fb, nb, C, np, ns, rad, Kpad, bufG, tot);
    int M = BB * np * ns;
    int c1 = saDims[sidx][1], c2 = saDims[sidx][2], c3 = saDims[sidx][3];
    gemm(bufG, saWb[sidx][0], saBias[sidx][0], bufA, nullptr, M, K, Kpad, c1);
    gemm(bufA, saWb[sidx][1], saBias[sidx][1], bufB, nullptr, M, c1, c1, c2);
    gemm(bufB, saWb[sidx][2], saBias[sidx][2], bufA, nullptr, M, c2, c2, c3);
    tot = BB * np * c3;
    k_maxpool<<<g256(tot), 256, 0, stream>>>(bufA, ns, c3, fout, tot);
  };

  // ---- stage 0 ----
  int tot = BB * NN;
  k_extract_xyz<<<g256(tot), 256, 0, stream>>>(d_pc, xyz0, tot);
  k_paint<<<g256(tot), 256, 0, stream>>>(d_pc, painted, tot);
  k_umb<<<g256(tot), 256, 0, stream>>>(d_rs, umbW, umbB, feats0, tot);

  // ---- SA1 pass 1 ----
  run_sa(xyz0, painted, NN, xyz0, feats0, NN, 10, 1024, 0.2f, 64, 1.0f, 0,
         sa1_i1, c11, sa1_p1, sa1_f1);

  // ---- remove sampled ----
  k_zero_i<<<g256(BB * NN), 256, 0, stream>>>(mask, BB * NN);
  k_scatter1<<<g256(BB * 1024), 256, 0, stream>>>(mask, sa1_i1, 1024, NN, BB * 1024);
  k_compact<<<BB, 1024, 0, stream>>>(mask, NN, NN - 1024, order);
  tot = BB * 15360 * 3;
  k_gather_rows_f<<<g256(tot), 256, 0, stream>>>(xyz0, NN, order, 15360, 3, nxyz, tot);
  tot = BB * 15360;
  k_gather_i<<<g256(tot), 256, 0, stream>>>(painted, NN, order, 15360, npnt, tot);
  tot = BB * 15360 * 10;
  k_gather_rows_f<<<g256(tot), 256, 0, stream>>>(feats0, NN, order, 15360, 10, nfeat, tot);

  // ---- SA1 pass 2 (ball over full set) ----
  run_sa(nxyz, npnt, 15360, xyz0, feats0, NN, 10, 1024, 0.2f, 64, 2.0f, 0,
         sa1_i2, c12, sa1_p2, sa1_f2);

  // concat sa1 (xyz straight into d_out)
  tot = BB * 1024 * 3;
  k_copy_rows_f<<<g256(tot), 256, 0, stream>>>(c11, out_sa1xyz, 1024, 3, 2048, 0, tot);
  k_copy_rows_f<<<g256(tot), 256, 0, stream>>>(c12, out_sa1xyz, 1024, 3, 2048, 1024, tot);
  tot = BB * 1024 * 128;
  k_copy_rows_f<<<g256(tot), 256, 0, stream>>>(sa1_f1, sa1_f, 1024, 128, 2048, 0, tot);
  k_copy_rows_f<<<g256(tot), 256, 0, stream>>>(sa1_f2, sa1_f, 1024, 128, 2048, 1024, tot);

  // ---- SA2 ----
  run_sa(c11, sa1_p1, 1024, c11, sa1_f1, 1024, 128, 512, 0.4f, 32, 1.0f, 1,
         sa2_i1, c21, sa2_p1, sa2_f1);
  run_sa(c12, sa1_p2, 1024, out_sa1xyz, sa1_f, 2048, 128, 512, 0.4f, 32, 2.0f, 1,
         sa2_i2, c22, sa2_p2, sa2_f2);
  tot = BB * 512 * 3;
  k_copy_rows_f<<<g256(tot), 256, 0, stream>>>(c21, out_sa2xyz, 512, 3, 1024, 0, tot);
  k_copy_rows_f<<<g256(tot), 256, 0, stream>>>(c22, out_sa2xyz, 512, 3, 1024, 512, tot);
  tot = BB * 512 * 256;
  k_copy_rows_f<<<g256(tot), 256, 0, stream>>>(sa2_f1, sa2_f, 512, 256, 1024, 0, tot);
  k_copy_rows_f<<<g256(tot), 256, 0, stream>>>(sa2_f2, sa2_f, 512, 256, 1024, 512, tot);

  // ---- SA3 ----
  run_sa(c21, sa2_p1, 512, c21, sa2_f1, 512, 256, 256, 0.8f, 16, 1.0f, 2,
         sa3_i1, c31, sa3_p1, sa3_f1);
  run_sa(c22, sa2_p2, 512, out_sa2xyz, sa2_f, 1024, 256, 256, 0.8f, 16, 2.0f, 2,
         sa3_i2, c32, sa3_p2, sa3_f2);
  tot = BB * 256 * 3;
  k_copy_rows_f<<<g256(tot), 256, 0, stream>>>(c31, sa3_xyz, 256, 3, 512, 0, tot);
  k_copy_rows_f<<<g256(tot), 256, 0, stream>>>(c32, sa3_xyz, 256, 3, 512, 256, tot);
  tot = BB * 256 * 256;
  k_copy_rows_f<<<g256(tot), 256, 0, stream>>>(sa3_f1, sa3_f, 256, 256, 512, 0, tot);
  k_copy_rows_f<<<g256(tot), 256, 0, stream>>>(sa3_f2, sa3_f, 256, 256, 512, 256, tot);
  tot = BB * 256;
  k_copy_i<<<g256(tot), 256, 0, stream>>>(sa3_p1, sa3_p, 256, 512, 0, tot);
  k_copy_i<<<g256(tot), 256, 0, stream>>>(sa3_p2, sa3_p, 256, 512, 256, tot);

  // ---- SA4 ----
  run_sa(sa3_xyz, sa3_p, 512, sa3_xyz, sa3_f, 512, 256, 256, 1.2f, 16, 2.0f, 3,
         sa4_i, c4, sa4_p, sa4_f);

  // ---- FP1 ----
  tot = BB * 512;
  k_fp_build<<<g256(tot), 256, 0, stream>>>(sa3_xyz, 512, c4, 256, sa3_f, sa4_f, 256, bufG, tot);
  gemm(bufG, fpWb[0][0], fpBias[0][0], bufA, nullptr, BB * 512, 512, 512, 256);
  gemm(bufA, fpWb[0][1], fpBias[0][1], bufB, fp1_f, BB * 512, 256, 256, 256);

  // ---- FP2 (final f32 directly into d_out) ----
  tot = BB * 1024;
  k_fp_build<<<g256(tot), 256, 0, stream>>>(out_sa2xyz, 1024, sa3_xyz, 512, sa2_f, fp1_f, 256, bufG, tot);
  gemm(bufG, fpWb[1][0], fpBias[1][0], bufA, nullptr, BB * 1024, 512, 512, 256);
  gemm(bufA, fpWb[1][1], fpBias[1][1], bufB, out_fp2f, BB * 1024, 256, 256, 256);

  // ---- index bookkeeping ----
  tot = BB * 512;
  k_gather_i<<<g256(tot), 256, 0, stream>>>(sa1_i1, 1024, sa2_i1, 512, seed1, tot);
  k_gather_i<<<g256(tot), 256, 0, stream>>>(sa1_i2, 1024, sa2_i2, 512, sa12, tot);
  k_gather_i<<<g256(tot), 256, 0, stream>>>(order, 15360, sa12, 512, seed2, tot);
  tot = BB * 1024;
  k_gather_i<<<g256(tot), 256, 0, stream>>>(order, 15360, sa1_i2, 1024, i2orig, tot);

  tot = BB * 1024;
  k_copy_i<<<g256(tot), 256, 0, stream>>>(sa1_i1, sa1_inds_i, 1024, 2048, 0, tot);
  k_copy_i<<<g256(tot), 256, 0, stream>>>(i2orig, sa1_inds_i, 1024, 2048, 1024, tot);
  tot = BB * 512;
  k_copy_i<<<g256(tot), 256, 0, stream>>>(seed1, fp2_inds_i, 512, 1024, 0, tot);
  k_copy_i<<<g256(tot), 256, 0, stream>>>(seed2, fp2_inds_i, 512, 1024, 512, tot);

  k_i2f<<<g256(BB * 2048), 256, 0, stream>>>(sa1_inds_i, out_sa1inds, BB * 2048);
  k_i2f<<<g256(BB * 1024), 256, 0, stream>>>(fp2_inds_i, out_fp2inds, BB * 1024);
}
